// FFN_833223656414
// MI455X (gfx1250) — matrix operand order repaired and checked by reading, not compiled, not run on hardware
//
#include <hip/hip_runtime.h>

// ---------------- common types / helpers ----------------
typedef __attribute__((ext_vector_type(16))) __bf16 v16bf;
typedef __attribute__((ext_vector_type(8)))  float  v8f;

union FragBF { v16bf v; uint4 u[2]; };

static __device__ __forceinline__ unsigned short f2bf(float f) {
    unsigned int u = __float_as_uint(f);
    unsigned int r = (u + 0x7FFFu + ((u >> 16) & 1u)) >> 16;   // RNE
    return (unsigned short)r;
}

// low 32 bits of the generic (flat) address of a __shared__ object == LDS byte offset
static __device__ __forceinline__ unsigned lds_off(const void* p) {
    return (unsigned)(unsigned long long)(uintptr_t)p;
}

// async global -> LDS 16B copy (GV mode), tracked by ASYNCcnt; bypasses VGPR data path
static __device__ __forceinline__ void async_cp16(unsigned lds, const void* g) {
    asm volatile("global_load_async_to_lds_b128 %0, %1, off"
                 :: "v"(lds), "v"((unsigned long long)(uintptr_t)g)
                 : "memory");
}
static __device__ __forceinline__ void wait_async_le4() {
    asm volatile("s_wait_asynccnt 4" ::: "memory");
}
static __device__ __forceinline__ void wait_async_0() {
    asm volatile("s_wait_asynccnt 0" ::: "memory");
}

constexpr int Dc = 512;
constexpr int Mc = 1024;
constexpr int BM = 128, BN = 128, BK = 32;
constexpr int LSTR = 40;            // LDS row stride in halfwords (80B, 16B aligned, bank-spread)

// ---------------- weight transpose + bf16 convert:  Wt[n][k] = bf16(W[k][n]) ----------------
__global__ void wprep_kernel(const float* __restrict__ W, unsigned short* __restrict__ Wt,
                             int K, int N) {
    int k = blockIdx.x * blockDim.x + threadIdx.x;
    int n = blockIdx.y * blockDim.y + threadIdx.y;
    if (k < K && n < N)
        Wt[(long long)n * K + k] = f2bf(W[(long long)k * N + n]);
}

// ---------------- per-token normalization ----------------
__global__ __launch_bounds__(256) void norm_kernel(
    const float* __restrict__ x0, const float* __restrict__ x1,
    const float* __restrict__ g0, const float* __restrict__ b0, const float* __restrict__ g1,
    unsigned short* __restrict__ s0b, unsigned short* __restrict__ s1b,
    float* __restrict__ stats, long long ntok)
{
    const int t   = blockIdx.x;
    const int tid = threadIdx.x;
    const float* x0r = x0 + (long long)t * Dc;
    const float* x1r = x1 + (long long)t * Dc * 3;

    float s = 0.f, sq = 0.f, sq1 = 0.f;
    float v0[2], v1[6];
#pragma unroll
    for (int i = 0; i < 2; i++) { float v = x0r[tid + i * 256]; v0[i] = v; s += v; sq += v * v; }
#pragma unroll
    for (int i = 0; i < 6; i++) { float v = x1r[tid + i * 256]; v1[i] = v; sq1 += v * v; }

#pragma unroll
    for (int off = 16; off > 0; off >>= 1) {
        s   += __shfl_xor(s,   off, 32);
        sq  += __shfl_xor(sq,  off, 32);
        sq1 += __shfl_xor(sq1, off, 32);
    }
    __shared__ float r0[8], r1[8], r2[8];
    __shared__ float fmu, frstd, frinv;
    int lane = tid & 31, wid = tid >> 5;
    if (lane == 0) { r0[wid] = s; r1[wid] = sq; r2[wid] = sq1; }
    __syncthreads();
    if (tid == 0) {
        float S = 0.f, SQ = 0.f, SQ1 = 0.f;
        for (int i = 0; i < 8; i++) { S += r0[i]; SQ += r1[i]; SQ1 += r2[i]; }
        float mu   = S * (1.f / 512.f);
        float var  = SQ * (1.f / 512.f) - mu * mu;
        float rstd = rsqrtf(var + 1e-5f);
        float rinv = rsqrtf(SQ1 * (1.f / 512.f) + 1e-5f);
        fmu = mu; frstd = rstd; frinv = rinv;
        stats[t * 3 + 0] = mu; stats[t * 3 + 1] = rstd; stats[t * 3 + 2] = rinv;
    }
    __syncthreads();
    float mu = fmu, rstd = frstd, rinv = frinv;
#pragma unroll
    for (int i = 0; i < 2; i++) {
        int d = tid + i * 256;
        s0b[(long long)t * Dc + d] = f2bf((v0[i] - mu) * rstd * g0[d] + b0[d]);
    }
#pragma unroll
    for (int i = 0; i < 6; i++) {
        int e = tid + i * 256;           // e = d*3 + comp  (x1 is [...,D,3])
        int d = e / 3, comp = e % 3;
        s1b[(long long)comp * (ntok * Dc) + (long long)t * Dc + d] = f2bf(v1[i] * rinv * g1[d]);
    }
}

// ---------------- fused WMMA GEMM with async double-buffered LDS staging ----------------
struct GP {
    const unsigned short* A;   long long aPlane; int lda;
    const unsigned short* Bt;  int K;
    const float* bias;
    const float* gate;         // mode1 in
    float*       gateOut;      // mode0 out
    unsigned short* bfOut;     long long bfPlane;
    float*       fOut;
    const float* x;            // x0 (mode2) / x1 (mode3)
    const float* stats;
    const float* lng; const float* lnb;
};

// MODE 0: h0 = s0*W1s + b1s ; cols<M -> a0 = bf16(silu), cols>=M -> gate = sigmoid (f32)
// MODE 1: a1 = bf16( (s1_z*W1v) * gate )
// MODE 2: y0 = a0*W2s + b2s + skip0(x0,stats)         -> f32 out
// MODE 3: y1_z = (a1_z*W2v) + skip1(x1,stats)         -> f32 out, stride-3 interleave
template<int MODE>
__global__ __launch_bounds__(256) void gemm_kernel(GP p) {
    __shared__ unsigned short lA[2][BM * LSTR];
    __shared__ unsigned short lB[2][BN * LSTR];

    const int tid  = threadIdx.x;
    const int lane = tid & 31, wid = tid >> 5;
    const int wm = wid & 1, wn = wid >> 1;         // waves: 2 (rows) x 4 (cols)
    const int khalf = lane >> 4, l15 = lane & 15;
    const int col0 = blockIdx.x * BN;
    const int row0 = blockIdx.y * BM;
    const int z    = blockIdx.z;
    const unsigned short* A = p.A + (long long)z * p.aPlane;
    const int lda = p.lda, K = p.K;
    const int nk = K / BK;

    v8f acc[4][2] = {};

    // per-thread staging coordinates: 512 x 16B chunks per tile, 2 per thread
    const int r0s = (tid      ) >> 2, q0 = (tid      ) & 3;
    const int r1s = (tid + 256) >> 2, q1 = (tid + 256) & 3;

    const unsigned short* gA0 = &A[(long long)(row0 + r0s) * lda + q0 * 8];
    const unsigned short* gA1 = &A[(long long)(row0 + r1s) * lda + q1 * 8];
    const unsigned short* gB0 = &p.Bt[(long long)(col0 + r0s) * K + q0 * 8];
    const unsigned short* gB1 = &p.Bt[(long long)(col0 + r1s) * K + q1 * 8];

    unsigned la0[2], la1[2], lb0[2], lb1[2];
#pragma unroll
    for (int b = 0; b < 2; b++) {
        la0[b] = lds_off(&lA[b][r0s * LSTR + q0 * 8]);
        la1[b] = lds_off(&lA[b][r1s * LSTR + q1 * 8]);
        lb0[b] = lds_off(&lB[b][r0s * LSTR + q0 * 8]);
        lb1[b] = lds_off(&lB[b][r1s * LSTR + q1 * 8]);
    }

    // one K-step of fragment loads + 8 WMMAs from buffer `cur`
    auto computeTile = [&](int cur) {
        FragBF a[4], b[2];
#pragma unroll
        for (int mt = 0; mt < 4; mt++) {
            // A 16x32 bf16: lane m=l15, vgpr0-3 = K khalf*8..+7, vgpr4-7 = K 16+khalf*8..+7
            const unsigned short* s = &lA[cur][(wm * 64 + mt * 16 + l15) * LSTR + khalf * 8];
            a[mt].u[0] = *(const uint4*)s;
            a[mt].u[1] = *(const uint4*)(s + 16);
        }
#pragma unroll
        for (int nt = 0; nt < 2; nt++) {
            // B 32x16 bf16 (column-major per lane): lane n=l15, vgpr0-3 = K khalf*8..+7, vgpr4-7 = K 16+khalf*8..+7
            const unsigned short* s = &lB[cur][(wn * 32 + nt * 16 + l15) * LSTR + khalf * 8];
            b[nt].u[0] = *(const uint4*)s;
            b[nt].u[1] = *(const uint4*)(s + 16);
        }
#pragma unroll
        for (int mt = 0; mt < 4; mt++)
#pragma unroll
            for (int nt = 0; nt < 2; nt++)
                acc[mt][nt] = __builtin_amdgcn_wmma_f32_16x16x32_bf16(
                    false, a[mt].v, false, b[nt].v, (short)0, acc[mt][nt], false, false);
    };

    // prologue: stage tile 0 into buffer 0
    async_cp16(la0[0], gA0);
    async_cp16(la1[0], gA1);
    async_cp16(lb0[0], gB0);
    async_cp16(lb1[0], gB1);

    // steady state: branch-free body (issue next -> wait current -> compute)
    for (int ks = 0; ks < nk - 1; ks++) {
        const int cur = ks & 1;
        const long long ko = (long long)(ks + 1) * BK;
        async_cp16(la0[cur ^ 1], gA0 + ko);
        async_cp16(la1[cur ^ 1], gA1 + ko);
        async_cp16(lb0[cur ^ 1], gB0 + ko);
        async_cp16(lb1[cur ^ 1], gB1 + ko);
        wait_async_le4();                  // current tile's 4 copies complete (in-order)
        __syncthreads();
        computeTile(cur);
        __syncthreads();                   // all waves done with buffer `cur` before refill
    }
    // peeled last tile
    wait_async_0();
    __syncthreads();
    computeTile((nk - 1) & 1);

    // ---- epilogue: C/D layout -> m = vgpr + 8*khalf, n = l15 ----
#pragma unroll
    for (int mt = 0; mt < 4; mt++) {
#pragma unroll
        for (int nt = 0; nt < 2; nt++) {
            const int c = col0 + wn * 32 + nt * 16 + l15;
#pragma unroll
            for (int j = 0; j < 8; j++) {
                const int t = row0 + wm * 64 + mt * 16 + khalf * 8 + j;
                float v = acc[mt][nt][j];
                if (MODE == 0) {
                    v += p.bias[c];
                    float sg = 1.f / (1.f + __expf(-v));
                    if (c < Mc) p.bfOut[(long long)t * Mc + c] = f2bf(v * sg);
                    else        p.gateOut[(long long)t * Mc + (c - Mc)] = sg;
                } else if (MODE == 1) {
                    v *= p.gate[(long long)t * Mc + c];
                    p.bfOut[(long long)z * p.bfPlane + (long long)t * Mc + c] = f2bf(v);
                } else if (MODE == 2) {
                    v += p.bias[c];
                    float mu = p.stats[t * 3], rstd = p.stats[t * 3 + 1];
                    float skip = (p.x[(long long)t * Dc + c] - mu) * rstd * p.lng[c] + p.lnb[c];
                    p.fOut[(long long)t * Dc + c] = v + skip;
                } else {
                    float rinv = p.stats[t * 3 + 2];
                    float skip = p.x[((long long)t * Dc + c) * 3 + z] * rinv * p.lng[c];
                    p.fOut[((long long)t * Dc + c) * 3 + z] = v + skip;
                }
            }
        }
    }
}

// ---------------- host wiring ----------------
extern "C" void kernel_launch(void* const* d_in, const int* in_sizes, int n_in,
                              void* d_out, int out_size, void* d_ws, size_t ws_size,
                              hipStream_t stream) {
    const float* x0  = (const float*)d_in[0];
    const float* x1  = (const float*)d_in[1];
    const float* g0  = (const float*)d_in[2];
    const float* b0  = (const float*)d_in[3];
    const float* g1  = (const float*)d_in[4];
    const float* W1s = (const float*)d_in[5];
    const float* b1s = (const float*)d_in[6];
    const float* W1v = (const float*)d_in[7];
    const float* W2s = (const float*)d_in[8];
    const float* b2s = (const float*)d_in[9];
    const float* W2v = (const float*)d_in[10];
    float* out = (float*)d_out;

    const long long ntok = (long long)in_sizes[0] / Dc;   // 16384

    char* ws = (char*)d_ws;
    auto alloc = [&](size_t bytes) -> char* {
        char* p = ws; ws += (bytes + 255) & ~(size_t)255; return p;
    };
    unsigned short* W1st = (unsigned short*)alloc((size_t)2 * Mc * Dc * 2);   // [2M][D]
    unsigned short* W1vt = (unsigned short*)alloc((size_t)Mc * Dc * 2);       // [M][D]
    unsigned short* W2st = (unsigned short*)alloc((size_t)Dc * Mc * 2);       // [D][M]
    unsigned short* W2vt = (unsigned short*)alloc((size_t)Dc * Mc * 2);       // [D][M]
    unsigned short* s0b  = (unsigned short*)alloc((size_t)ntok * Dc * 2);
    unsigned short* s1b  = (unsigned short*)alloc((size_t)3 * ntok * Dc * 2);
    unsigned short* a0b  = (unsigned short*)alloc((size_t)ntok * Mc * 2);
    unsigned short* a1b  = (unsigned short*)alloc((size_t)3 * ntok * Mc * 2);
    float*          gate = (float*)alloc((size_t)ntok * Mc * 4);
    float*          stats= (float*)alloc((size_t)ntok * 3 * 4);

    // 1) weights -> bf16 transposed [N][K]
    {
        dim3 blk(32, 8);
        wprep_kernel<<<dim3((Dc + 31) / 32, (2 * Mc + 7) / 8), blk, 0, stream>>>(W1s, W1st, Dc, 2 * Mc);
        wprep_kernel<<<dim3((Dc + 31) / 32, (Mc + 7) / 8),     blk, 0, stream>>>(W1v, W1vt, Dc, Mc);
        wprep_kernel<<<dim3((Mc + 31) / 32, (Dc + 7) / 8),     blk, 0, stream>>>(W2s, W2st, Mc, Dc);
        wprep_kernel<<<dim3((Mc + 31) / 32, (Dc + 7) / 8),     blk, 0, stream>>>(W2v, W2vt, Mc, Dc);
    }
    // 2) normalization
    norm_kernel<<<dim3((unsigned)ntok), 256, 0, stream>>>(x0, x1, g0, b0, g1, s0b, s1b, stats, ntok);

    const unsigned tokBlocks = (unsigned)(ntok / BM);

    // 3) FF1 scalar: h0 -> a0 (bf16 silu) + gate (f32 sigmoid)
    {
        GP p{}; p.A = s0b; p.aPlane = 0; p.lda = Dc; p.Bt = W1st; p.K = Dc;
        p.bias = b1s; p.gateOut = gate; p.bfOut = a0b;
        gemm_kernel<0><<<dim3((2 * Mc) / BN, tokBlocks, 1), 256, 0, stream>>>(p);
    }
    // 4) FF1 vector (x3): a1 = bf16(h1 * gate)
    {
        GP p{}; p.A = s1b; p.aPlane = ntok * Dc; p.lda = Dc; p.Bt = W1vt; p.K = Dc;
        p.gate = gate; p.bfOut = a1b; p.bfPlane = ntok * Mc;
        gemm_kernel<1><<<dim3(Mc / BN, tokBlocks, 3), 256, 0, stream>>>(p);
    }
    // 5) FF2 scalar: y0 = a0*W2s + b2s + skip0
    {
        GP p{}; p.A = a0b; p.aPlane = 0; p.lda = Mc; p.Bt = W2st; p.K = Mc;
        p.bias = b2s; p.fOut = out; p.x = x0; p.stats = stats; p.lng = g0; p.lnb = b0;
        gemm_kernel<2><<<dim3(Dc / BN, tokBlocks, 1), 256, 0, stream>>>(p);
    }
    // 6) FF2 vector (x3): y1 = a1*W2v + skip1, interleaved [...,D,3]
    {
        GP p{}; p.A = a1b; p.aPlane = ntok * Mc; p.lda = Mc; p.Bt = W2vt; p.K = Mc;
        p.fOut = out + (long long)ntok * Dc; p.x = x1; p.stats = stats; p.lng = g1;
        gemm_kernel<3><<<dim3(Dc / BN, tokBlocks, 3), 256, 0, stream>>>(p);
    }
}
